// PCENetwork_34883724378593
// MI455X (gfx1250) — compile-verified, matrix operand order checked by reading, not run on hardware
//
#include <hip/hip_runtime.h>
#include <hip/hip_bf16.h>

// ---------------------------------------------------------------------------
// MoE patch conv for MI455X (gfx1250, wave32, WMMA):
//   out = conv(x, sum_e s_e * W_e) + sum_e s_e * b_e   (expert mixing commutes
//   with the linear conv -> 8x FLOP cut). Conv = implicit GEMM over 9 taps,
//   K = C = 32 matches v_wmma_f32_16x16x32_bf16 exactly.
// LDS channel order is pre-permuted to the WMMA 16-bit fragment K order
// ([0..7,16..23 | 8..15,24..31]) so every fragment is two ds_load_b128.
// ---------------------------------------------------------------------------

typedef __attribute__((ext_vector_type(16))) __bf16        v16bf;
typedef __attribute__((ext_vector_type(8)))  float         v8f;
typedef __attribute__((ext_vector_type(4)))  unsigned int  v4u;

union Frag {
    v16bf v;
    v4u   q[2];   // 2 x 16B = 16 bf16
};

__device__ __forceinline__ unsigned short f2bf(float f) {
    unsigned int u = __float_as_uint(f);
    unsigned int lsb = (u >> 16) & 1u;
    u += 0x7fffu + lsb;                 // round-to-nearest-even
    return (unsigned short)(u >> 16);
}

// Involutive channel permutation: swap 8-blocks 1 and 2, so that a half-wave's
// 16 fragment K-values are contiguous in LDS.
__device__ __forceinline__ int kperm(int c) {
    int blk = c >> 3;
    return (blk == 1) ? (c + 8) : (blk == 2) ? (c - 8) : c;
}

// ------------------------- dimensions (compile-time) ------------------------
#define BATCH 8
#define CIN   32
#define HW    256
#define NEXP  8
#define COUT  64
#define PH    32          // patch size
#define PHW   34          // halo'd patch size
#define NPATCH 512        // B * 8 * 8 patches

// ---------------------------------------------------------------------------
// Kernel 1: router. One block per patch: mean-pool -> 32x8 linear -> softmax.
// ---------------------------------------------------------------------------
__global__ __launch_bounds__(256)
void moe_router_kernel(const float* __restrict__ X,
                       const float* __restrict__ rw,   // (C,E)
                       const float* __restrict__ rb,   // (E)
                       float* __restrict__ scores)     // (NPATCH, E)
{
    int p = blockIdx.x;                 // 0..511
    int b = p >> 6, pid = p & 63, gy = pid >> 3, gx = pid & 7;

    __shared__ float red[256];
    __shared__ float feats[CIN];
    __shared__ float logit[NEXP];

    int t = threadIdx.x;
    int c = t & 31, slice = t >> 5;     // 8 slices x 128 pixels

    const float* base =
        X + (((size_t)(b * CIN + c)) * HW + gy * PH) * HW + gx * PH;
    float s = 0.f;
    for (int i = 0; i < 128; ++i) {
        int pix = slice * 128 + i;
        int yy = pix >> 5, xx = pix & 31;
        s += base[yy * HW + xx];
    }
    red[t] = s;
    __syncthreads();

    if (t < CIN) {
        float acc = 0.f;
        #pragma unroll
        for (int k = 0; k < 8; ++k) acc += red[t + 32 * k];
        feats[t] = acc * (1.0f / (PH * PH));
    }
    __syncthreads();

    if (t < NEXP) {
        float l = rb[t];
        #pragma unroll
        for (int cc = 0; cc < CIN; ++cc) l += feats[cc] * rw[cc * NEXP + t];
        logit[t] = l;
    }
    __syncthreads();

    if (t == 0) {
        float mx = logit[0];
        #pragma unroll
        for (int e = 1; e < NEXP; ++e) mx = fmaxf(mx, logit[e]);
        float ex[NEXP], den = 0.f;
        #pragma unroll
        for (int e = 0; e < NEXP; ++e) { ex[e] = __expf(logit[e] - mx); den += ex[e]; }
        float inv = 1.0f / den;
        #pragma unroll
        for (int e = 0; e < NEXP; ++e) scores[p * NEXP + e] = ex[e] * inv;
    }
}

// ---------------------------------------------------------------------------
// Kernel 2: per-patch mixed-weight conv via WMMA implicit GEMM.
// One block (8 wave32) per patch; wave w owns co-tile (w&3), pixel-half (w>>2).
// Dynamic LDS ~111 KB (of 320 KB / WGP):
//   inS : bf16 [PHW*PHW][32]  halo input, channel-last, kperm'd
//   WbS : bf16 [9][64][32]    mixed weights, kperm'd channels contiguous
//   bias: f32  [64]           mixed bias
// ---------------------------------------------------------------------------
__global__ __launch_bounds__(256)
void moe_conv_wmma_kernel(const float* __restrict__ X,
                          const float* __restrict__ cw,     // (E,CO,C,3,3)
                          const float* __restrict__ cb,     // (E,CO)
                          const float* __restrict__ scores, // (NPATCH,E)
                          float* __restrict__ out)          // (B,CO,256,256)
{
    int p = blockIdx.x;
    int b = p >> 6, pid = p & 63, gy = pid >> 3, gx = pid & 7;

    extern __shared__ unsigned char smem[];
    unsigned short* inS   = (unsigned short*)smem;            // PHW*PHW*32
    unsigned short* WbS   = inS + PHW * PHW * CIN;            // 9*64*32 (16B-aligned)
    float*          biasE = (float*)(WbS + 9 * COUT * CIN);   // 64
    float*          sS    = biasE + COUT;                     // 8

    int t = threadIdx.x;

    if (t < NEXP) sS[t] = scores[p * NEXP + t];
    __syncthreads();

    float sc[NEXP];
    #pragma unroll
    for (int e = 0; e < NEXP; ++e) sc[e] = sS[e];

    // ---- mix expert weights into LDS (bf16), layout [tap][co][kperm(c)] ----
    for (int idx = t; idx < 9 * COUT * CIN; idx += 256) {
        int tap = idx >> 11;            // / (64*32)
        int rem = idx & 2047;
        int co  = rem >> 5;
        int ccc = kperm(rem & 31);      // channel stored at this position
        int ky = tap / 3, kx = tap % 3;
        float w = 0.f;
        #pragma unroll
        for (int e = 0; e < NEXP; ++e)
            w += sc[e] * cw[((((size_t)(e * COUT + co)) * CIN + ccc) * 3 + ky) * 3 + kx];
        WbS[idx] = f2bf(w);
    }
    if (t < COUT) {
        float bs = 0.f;
        #pragma unroll
        for (int e = 0; e < NEXP; ++e) bs += sc[e] * cb[e * COUT + t];
        biasE[t] = bs;
    }

    // ---- stage halo'd input patch as bf16, channel-last + kperm ------------
    for (int ccc = 0; ccc < CIN; ++ccc) {
        const float* src = X + ((size_t)(b * CIN + ccc)) * HW * HW;
        int dstc = kperm(ccc);
        for (int pix = t; pix < PHW * PHW; pix += 256) {
            int yy = pix / PHW, xx = pix % PHW;
            int row = gy * PH + yy - 1;
            int col = gx * PH + xx - 1;
            float v = 0.f;
            if (row >= 0 && row < HW && col >= 0 && col < HW)
                v = src[row * HW + col];
            inS[pix * CIN + dstc] = f2bf(v);
        }
    }
    __syncthreads();

    // ---- WMMA implicit GEMM ------------------------------------------------
    int lane = t & 31, wave = t >> 5;
    int n    = lane & 15;          // M (for A) / N (for B) position
    int half = lane >> 4;          // selects which 16-element K run
    int ct   = wave & 3;           // this wave's co-tile (co base = ct*16)

    // Hoist all 9 A fragments (one co-tile) and the bias vector.
    Frag aF[9];
    #pragma unroll
    for (int tap = 0; tap < 9; ++tap) {
        const unsigned short* ap =
            WbS + (tap * COUT + ct * 16 + n) * CIN + half * 16;
        aF[tap].q[0] = *(const v4u*)ap;
        aF[tap].q[1] = *(const v4u*)(ap + 8);
    }
    v8f biasv;
    #pragma unroll
    for (int r = 0; r < 8; ++r) biasv[r] = biasE[ct * 16 + r + half * 8];

    for (int pt = (wave >> 2); pt < 64; pt += 2) {   // 32 pixel tiles / wave
        int y  = pt >> 1;
        int xh = (pt & 1) << 4;

        v8f acc = biasv;
        const unsigned short* bp0 =
            inS + (y * PHW + xh + n) * CIN + half * 16;

        #pragma unroll
        for (int tap = 0; tap < 9; ++tap) {
            int dy = tap / 3, dx = tap % 3;
            const unsigned short* bp = bp0 + (dy * PHW + dx) * CIN;  // const offs
            Frag bF;
            bF.q[0] = *(const v4u*)bp;
            bF.q[1] = *(const v4u*)(bp + 8);
            acc = __builtin_amdgcn_wmma_f32_16x16x32_bf16(
                false, aF[tap].v, false, bF.v, (short)0, acc, false, false);
        }

        int xg = gx * PH + xh + n;
        int yg = gy * PH + y;
        #pragma unroll
        for (int r = 0; r < 8; ++r) {
            int co = ct * 16 + r + half * 8;
            out[(((size_t)(b * COUT + co)) * HW + yg) * HW + xg] = acc[r];
        }
    }
}

// ---------------------------------------------------------------------------
extern "C" void kernel_launch(void* const* d_in, const int* in_sizes, int n_in,
                              void* d_out, int out_size, void* d_ws, size_t ws_size,
                              hipStream_t stream) {
    const float* X  = (const float*)d_in[0];   // (B,C,H,W)
    const float* cw = (const float*)d_in[1];   // (E,CO,C,3,3)
    const float* cb = (const float*)d_in[2];   // (E,CO)
    const float* rw = (const float*)d_in[3];   // (C,E)
    const float* rb = (const float*)d_in[4];   // (E)
    float* out = (float*)d_out;
    float* scores = (float*)d_ws;              // NPATCH*8 floats

    moe_router_kernel<<<NPATCH, 256, 0, stream>>>(X, rw, rb, scores);

    size_t smem = (size_t)PHW * PHW * CIN * sizeof(unsigned short)   // input
                + (size_t)9 * COUT * CIN * sizeof(unsigned short)    // weights
                + (size_t)COUT * sizeof(float)                       // bias
                + (size_t)NEXP * sizeof(float);                      // scores
    moe_conv_wmma_kernel<<<NPATCH, 256, smem, stream>>>(X, cw, cb, scores, out);
}
